// AdjacencyBasedLoss_27504970563869
// MI455X (gfx1250) — matrix operand order repaired and checked by reading, not compiled, not run on hardware
//
#include <hip/hip_runtime.h>

typedef __attribute__((ext_vector_type(16))) _Float16 v16h;
typedef __attribute__((ext_vector_type(8)))  _Float16 h8;
typedef __attribute__((ext_vector_type(8)))  float    v8f;
typedef __attribute__((ext_vector_type(4)))  int      v4i;

#define AS1 __attribute__((address_space(1)))
#define AS3 __attribute__((address_space(3)))

#define NN     8192
#define DD     256
#define MT     64
#define KT     32
#define KCHUNK 2048
#define NSTEPS (KCHUNK / KT)
#define AS_STRIDE 40    // 32 + 8 f16 pad
#define WS_STRIDE 264   // 256 + 8 f16 pad

#if __has_builtin(__builtin_amdgcn_global_load_async_to_lds_b128) && \
    __has_builtin(__builtin_amdgcn_s_wait_asynccnt)
#define HAVE_ASYNC_LDS 1
#else
#define HAVE_ASYNC_LDS 0
#endif

// ---------------- Kernel 1: row sums of A -> d_inv_sqrt; zero accumulators ----
__global__ __launch_bounds__(256) void k_rowsum(const float* __restrict__ A,
                                                float* __restrict__ dinv,
                                                float* __restrict__ u,
                                                float* __restrict__ Sacc) {
  __shared__ float red[256];
  const int row = blockIdx.x, t = threadIdx.x;
  float s = 0.f;
  const float* ar = A + (size_t)row * NN;
  for (int c = t; c < NN; c += 256) s += ar[c];
  red[t] = s; __syncthreads();
  for (int st = 128; st > 0; st >>= 1) { if (t < st) red[t] += red[t + st]; __syncthreads(); }
  if (t == 0) dinv[row] = rsqrtf(red[0] + 1e-10f);
  if (row == 0) { u[t] = 0.f; if (t == 0) *Sacc = 0.f; }
}

// -------- Kernel 2: W = d_i * Z_i / max(||Z_i||,1e-12) (f16); u += Z_n rows ---
__global__ __launch_bounds__(256) void k_normalize(const float* __restrict__ Z,
                                                   const float* __restrict__ dinv,
                                                   _Float16* __restrict__ Wbuf,
                                                   float* __restrict__ u) {
  __shared__ float red[256];
  const int row = blockIdx.x, t = threadIdx.x;
  const float z = Z[(size_t)row * DD + t];
  red[t] = z * z; __syncthreads();
  for (int st = 128; st > 0; st >>= 1) { if (t < st) red[t] += red[t + st]; __syncthreads(); }
  const float inv = 1.f / fmaxf(sqrtf(red[0]), 1e-12f);
  const float zn = z * inv;
  Wbuf[(size_t)row * DD + t] = (_Float16)(zn * dinv[row]);
  atomicAdd(&u[t], zn);
}

#if HAVE_ASYNC_LDS
__device__ __forceinline__ void async_row64(const _Float16* g, _Float16* l) {
  // copy 64 contiguous bytes global -> LDS as 4 async b128 transfers
  __builtin_amdgcn_global_load_async_to_lds_b128((AS1 v4i*)g, (AS3 v4i*)l, 0, 0);
  __builtin_amdgcn_global_load_async_to_lds_b128((AS1 v4i*)g, (AS3 v4i*)l, 16, 0);
  __builtin_amdgcn_global_load_async_to_lds_b128((AS1 v4i*)g, (AS3 v4i*)l, 32, 0);
  __builtin_amdgcn_global_load_async_to_lds_b128((AS1 v4i*)g, (AS3 v4i*)l, 48, 0);
}
#endif

// -------- Kernel 3: S += sum( W o (A @ W) ) over an M-tile x K-chunk ----------
__global__ __launch_bounds__(256) void k_gemm_reduce(const float* __restrict__ A,
                                                     const _Float16* __restrict__ Wbuf,
                                                     float* __restrict__ Sacc) {
  __shared__ _Float16 As[2][MT * AS_STRIDE];
  __shared__ _Float16 Ws[2][KT * WS_STRIDE];
  __shared__ float red[256];

  const int tid  = threadIdx.x;
  const int lane = tid & 31;
  const int wid  = tid >> 5;
  const int wave_m = wid >> 1;          // 0..3 -> 16-row slice
  const int wave_n = wid & 1;           // 0..1 -> 128-col half

  const int i0    = blockIdx.x * MT;
  const int kbase = blockIdx.y * KCHUNK;

  v8f c[8] = {};

  // A loader: 4 threads/row, 8 f32 each. W loader: 8 threads/row, 64B each.
  const int a_r = tid >> 2, a_c = (tid & 3) * 8;
  const int w_r = tid >> 3, w_c = (tid & 7) * 32;

  const float* a_gsrc = A + (size_t)(i0 + a_r) * NN + (kbase + a_c);

  // ---- issue W(0) copy toward LDS buf 0; stage A(0) in registers -------------
#if HAVE_ASYNC_LDS
  async_row64(Wbuf + (size_t)(kbase + w_r) * DD + w_c,
              &Ws[0][w_r * WS_STRIDE + w_c]);
#else
  h8 wreg[4];
  {
    const h8* wp = (const h8*)(Wbuf + (size_t)(kbase + w_r) * DD + w_c);
    wreg[0] = wp[0]; wreg[1] = wp[1]; wreg[2] = wp[2]; wreg[3] = wp[3];
  }
#endif
  float4 a0 = ((const float4*)a_gsrc)[0];
  float4 a1 = ((const float4*)a_gsrc)[1];

  for (int ks = 0; ks < NSTEPS; ++ks) {
    const int p = ks & 1;
    const bool has_next = (ks + 1 < NSTEPS);

    // ---- convert + store A(ks) into LDS buf p
    {
      h8 av;
      av[0]=(_Float16)a0.x; av[1]=(_Float16)a0.y; av[2]=(_Float16)a0.z; av[3]=(_Float16)a0.w;
      av[4]=(_Float16)a1.x; av[5]=(_Float16)a1.y; av[6]=(_Float16)a1.z; av[7]=(_Float16)a1.w;
      *(h8*)(&As[p][a_r * AS_STRIDE + a_c]) = av;
    }
    // ---- stage A(ks+1) in registers; prefetch A(ks+2)
    if (has_next) {
      const float4* an = (const float4*)(a_gsrc + (size_t)(ks + 1) * KT);
      a0 = an[0]; a1 = an[1];
      __builtin_prefetch(a_gsrc + (size_t)(ks + 2) * KT, 0, 3);
    }

#if HAVE_ASYNC_LDS
    __builtin_amdgcn_s_wait_asynccnt(0);   // my W(ks) pieces have landed in LDS
#endif
    __syncthreads();                        // all waves staged tile ks; reads of buf 1-p done

    // ---- kick off W(ks+1) into LDS buf 1-p (overlaps the WMMAs below)
    if (has_next) {
#if HAVE_ASYNC_LDS
      async_row64(Wbuf + (size_t)(kbase + (ks + 1) * KT + w_r) * DD + w_c,
                  &Ws[1 - p][w_r * WS_STRIDE + w_c]);
#else
      h8* wd = (h8*)(&Ws[1 - p][w_r * WS_STRIDE + w_c]);
      wd[0] = wreg[0]; wd[1] = wreg[1]; wd[2] = wreg[2]; wd[3] = wreg[3];
      if (ks + 2 < NSTEPS) {
        const h8* wp = (const h8*)(Wbuf + (size_t)(kbase + (ks + 2) * KT + w_r) * DD + w_c);
        wreg[0] = wp[0]; wreg[1] = wp[1]; wreg[2] = wp[2]; wreg[3] = wp[3];
      }
#endif
    }

    // ---- A fragment (16x32): lane<16 -> M=lane, K {0-7,16-23}; lane>=16 -> {8-15,24-31}
    const int arow = wave_m * 16 + (lane & 15);
    const int klo  = (lane < 16) ? 0 : 8;
    h8 alo = *(const h8*)(&As[p][arow * AS_STRIDE + klo]);
    h8 ahi = *(const h8*)(&As[p][arow * AS_STRIDE + klo + 16]);
    v16h afrag = __builtin_shufflevector(alo, ahi, 0,1,2,3,4,5,6,7,8,9,10,11,12,13,14,15);

#pragma unroll
    for (int t = 0; t < 8; ++t) {
      // ---- B fragment (32x16): lane -> N=lane%16, K {0-7,16-23} (lane<16) / {8-15,24-31}
      const int n0 = wave_n * 128 + t * 16;
      const int bn = n0 + (lane & 15);
      v16h bfrag;
#pragma unroll
      for (int s = 0; s < 8; ++s) {
        bfrag[s]     = Ws[p][(klo + s) * WS_STRIDE + bn];
        bfrag[8 + s] = Ws[p][(klo + 16 + s) * WS_STRIDE + bn];
      }
      c[t] = __builtin_amdgcn_wmma_f32_16x16x32_f16(
          false, afrag, false, bfrag, (short)0, c[t], false, false);
    }
  }

  // ---- fused reduction: S_partial = sum W[m,n] * Y[m,n] over this tile
  // C/D layout: VGPR r, lanes 0-15 -> M=r, N=lane ; lanes 16-31 -> M=r+8, N=lane-16
  float part = 0.f;
  const int mrow  = i0 + wave_m * 16 + 8 * (lane >> 4);
  const int ncol0 = wave_n * 128 + (lane & 15);
#pragma unroll
  for (int t = 0; t < 8; ++t) {
#pragma unroll
    for (int r = 0; r < 8; ++r) {
      part += (float)Wbuf[(size_t)(mrow + r) * DD + (ncol0 + t * 16)] * c[t][r];
    }
  }
  red[tid] = part; __syncthreads();
  for (int st = 128; st > 0; st >>= 1) { if (tid < st) red[tid] += red[tid + st]; __syncthreads(); }
  if (tid == 0) atomicAdd(Sacc, red[0]);
}

// ---------------- Kernel 4: T = ||u||^2 ; out = (-S, T - S) -------------------
__global__ __launch_bounds__(256) void k_finalize(const float* __restrict__ u,
                                                  const float* __restrict__ Sacc,
                                                  float* __restrict__ out) {
  __shared__ float red[256];
  const int t = threadIdx.x;
  const float v = u[t];
  red[t] = v * v; __syncthreads();
  for (int st = 128; st > 0; st >>= 1) { if (t < st) red[t] += red[t + st]; __syncthreads(); }
  if (t == 0) { const float S = *Sacc; out[0] = -S; out[1] = red[0] - S; }
}

extern "C" void kernel_launch(void* const* d_in, const int* in_sizes, int n_in,
                              void* d_out, int out_size, void* d_ws, size_t ws_size,
                              hipStream_t stream) {
  (void)in_sizes; (void)n_in; (void)out_size; (void)ws_size;
  const float* Z = (const float*)d_in[1];   // [8192,256]
  const float* A = (const float*)d_in[2];   // [8192,8192]
  float* out = (float*)d_out;               // (homo, hetero)

  char* ws = (char*)d_ws;
  float*    dinv = (float*)ws;                      // 8192 f32   (32 KB)
  float*    u    = (float*)(ws + 32768);            // 256 f32    (1 KB)
  float*    Sacc = (float*)(ws + 32768 + 1024);     // 1 f32
  _Float16* Wbuf = (_Float16*)(ws + 34816);         // 8192*256 f16 (4 MB)

  k_rowsum   <<<NN, 256, 0, stream>>>(A, dinv, u, Sacc);
  k_normalize<<<NN, 256, 0, stream>>>(Z, dinv, Wbuf, u);
  dim3 g3(NN / MT, NN / KCHUNK);                    // 128 x 4 = 512 workgroups
  k_gemm_reduce<<<g3, 256, 0, stream>>>(A, Wbuf, Sacc);
  k_finalize <<<1, 256, 0, stream>>>(u, Sacc, out);
}
